// NeRF_46471546142968
// MI455X (gfx1250) — matrix operand order repaired and checked by reading, not compiled, not run on hardware
//
#include <hip/hip_runtime.h>
#include <hip/hip_bf16.h>

typedef __attribute__((ext_vector_type(16))) _Float16 v16h;
typedef __attribute__((ext_vector_type(8)))  _Float16 v8h;
typedef __attribute__((ext_vector_type(8)))  float    v8f;

// ---------------- weight workspace layout (f16 elements) ----------------
// All weights stored transposed: Wt[N][Kpad], K padded to multiple of 32 and
// remapped so concatenated inputs line up with padded LDS segments.
#define OFF_D1 0                       // 256 x 64   (K: pos 63 -> pad 64)
#define OFF_D2 (OFF_D1 + 256*64)       // 256 x 256
#define OFF_D3 (OFF_D2 + 256*256)
#define OFF_D4 (OFF_D3 + 256*256)
#define OFF_E1 (OFF_D4 + 256*256)      // 256 x 320  (K: pos 63->64 | h 256)
#define OFF_E2 (OFF_E1 + 256*320)
#define OFF_E3 (OFF_E2 + 256*256)
#define OFF_E4 (OFF_E3 + 256*256)
#define OFF_E5 (OFF_E4 + 256*256)      // 272 x 256  (N: 257 -> pad 272)
#define OFF_C1 (OFF_E5 + 272*256)      // 256 x 288  (K: view 27->32 | feat 256)
#define OFF_C2 (OFF_C1 + 256*288)      // 16 x 256   (N: 3 -> pad 16)
#define WTOTAL (OFF_C2 + 16*256)

// ---------------- LDS layout (f16 elements) ----------------
// pos/view buffer: 128 rows, stride 104 (cols 0..63 pos(padded), 64..95 view(padded))
// bufA / bufB:     128 rows, stride 264 (row stride 528 B == 16 mod 256 -> no bank conflicts)
#define PSTR 104
#define HSTR 264
#define SM_POS 0
#define SM_A   (SM_POS + 128*PSTR)     // 13312
#define SM_B   (SM_A   + 128*HSTR)     // 47104
#define SM_F16_TOTAL (SM_B + 128*HSTR) // 80896 f16
#define SM_BYTES (SM_F16_TOTAL*2 + 128*4)   // + density f32 = 162304 bytes

// ------------------------------------------------------------------
// Weight convert/transpose/pad kernel.
// src layout: [Kin][Nsrc] row-major (fan_in x fan_out, fp32).
// dst layout: [Npad][Kpad] f16.  K remap: kp<seg0_len -> kp ; seg0_len<=kp<seg0_pad -> zero ;
//             kp>=seg0_pad -> seg0_len + (kp - seg0_pad).
// ------------------------------------------------------------------
__global__ void convert_weights(const float* __restrict__ src, int Kin, int Nsrc,
                                _Float16* __restrict__ dst, int Npad, int Kpad,
                                int seg0_len, int seg0_pad)
{
    int i = blockIdx.x * 256 + threadIdx.x;
    if (i >= Npad * Kpad) return;
    int n  = i / Kpad;
    int kp = i - n * Kpad;
    int ksrc;
    if (kp < seg0_len)      ksrc = kp;
    else if (kp < seg0_pad) ksrc = -1;
    else                    ksrc = seg0_len + (kp - seg0_pad);
    float v = 0.0f;
    if (n < Nsrc && ksrc >= 0 && ksrc < Kin) v = src[(size_t)ksrc * Nsrc + n];
    dst[(size_t)n * Kpad + kp] = (_Float16)v;
}

// ------------------------------------------------------------------
// GEMM chunk: one wave computes NT 16x16 output tiles (N tiles nt0..nt0+NT-1)
// for its own 16-row M tile, accumulating over (s0.ksteps + s1.ksteps) K-steps
// of 32. A comes from LDS, B from global (transposed f16 weights).
// ------------------------------------------------------------------
struct Seg { const _Float16* base; int stride; int ksteps; };

template <int NT, typename Epi>
__device__ __forceinline__ void gemm_chunk(Seg s0, Seg s1,
                                           const _Float16* __restrict__ wt, int kpad,
                                           int nt0, int mtile, int lane, Epi&& epi)
{
    v8f acc[NT];
#pragma unroll
    for (int t = 0; t < NT; ++t) acc[t] = (v8f)(0.0f);

    const int m  = lane & 15;
    const int hi = lane >> 4;

#pragma unroll
    for (int seg = 0; seg < 2; ++seg) {
        const Seg s   = seg ? s1 : s0;
        const int kw0 = seg ? s0.ksteps : 0;
        for (int kk = 0; kk < s.ksteps; ++kk) {
            // A fragment (16x32 f16): lane hi=0 holds K {0..15}, hi=1 holds {16..31}
            const _Float16* ap = s.base + (size_t)(mtile * 16 + m) * s.stride + kk * 32 + hi * 16;
            v8h alo = *(const v8h*)(ap);
            v8h ahi = *(const v8h*)(ap + 8);
            v16h a;
#pragma unroll
            for (int j = 0; j < 8; ++j) { a[j] = alo[j]; a[j + 8] = ahi[j]; }

            const int kw = (kw0 + kk) * 32;
#pragma unroll
            for (int t = 0; t < NT; ++t) {
                // B fragment (32x16 f16): lane holds column n, 16 contiguous K values
                const _Float16* bp = wt + (size_t)((nt0 + t) * 16 + m) * kpad + kw + hi * 16;
                v16h b = *(const v16h*)bp;
                acc[t] = __builtin_amdgcn_wmma_f32_16x16x32_f16(
                    false, a, false, b, (short)0, acc[t], false, false);
            }
        }
    }
#pragma unroll
    for (int t = 0; t < NT; ++t) epi(acc[t], nt0 + t);
}

// Standard 256-output layer: bias + optional ReLU, f16 result into LDS dst.
template <bool RELU>
__device__ __forceinline__ void layer_std(Seg s0, Seg s1,
                                          const _Float16* __restrict__ wt, int kpad,
                                          const float* __restrict__ bias,
                                          _Float16* dst, int mtile, int lane)
{
    const int m  = lane & 15;
    const int hi = lane >> 4;
    for (int nt0 = 0; nt0 < 16; nt0 += 8) {
        gemm_chunk<8>(s0, s1, wt, kpad, nt0, mtile, lane,
            [&](v8f acc, int nt) {
                float bv = bias[nt * 16 + m];
#pragma unroll
                for (int v = 0; v < 8; ++v) {
                    float val = acc[v] + bv;
                    if (RELU) val = fmaxf(val, 0.0f);
                    dst[(size_t)(mtile * 16 + v + 8 * hi) * HSTR + nt * 16 + m] = (_Float16)val;
                }
            });
    }
}

// ------------------------------------------------------------------
// Fused NeRF forward: 1 block = 128 rows, 8 waves, wave w owns rows [16w,16w+16)
// through all 11 layers (wave-private dataflow -> no inter-layer barriers).
// ------------------------------------------------------------------
__global__ __launch_bounds__(256) void nerf_fused(
    const float* __restrict__ x, const _Float16* __restrict__ wts,
    const float* bd1, const float* bd2, const float* bd3, const float* bd4,
    const float* be1, const float* be2, const float* be3, const float* be4,
    const float* be5, const float* bc1, const float* bc2,
    float* __restrict__ out)
{
    extern __shared__ _Float16 smem[];
    _Float16* sPos  = smem + SM_POS;                 // stride PSTR: pos cols 0..63, view cols 64..95
    _Float16* sA    = smem + SM_A;                   // stride HSTR
    _Float16* sB    = smem + SM_B;                   // stride HSTR
    float*    sDens = (float*)(smem + SM_F16_TOTAL); // 128 floats

    const int tid   = threadIdx.x;
    const int lane  = tid & 31;
    const int mtile = tid >> 5;          // wave id == M tile
    const int row0  = blockIdx.x * 128;
    const int m  = lane & 15;
    const int hi = lane >> 4;

    // ---- stage inputs: pos (63 -> 64 cols) and view (27 -> 32 cols) as f16 ----
    for (int i = tid; i < 128 * 64; i += 256) {
        int r = i >> 6, c = i & 63;
        float v = (c < 63) ? x[(size_t)(row0 + r) * 90 + c] : 0.0f;
        sPos[r * PSTR + c] = (_Float16)v;
    }
    for (int i = tid; i < 128 * 32; i += 256) {
        int r = i >> 5, c = i & 31;
        float v = (c < 27) ? x[(size_t)(row0 + r) * 90 + 63 + c] : 0.0f;
        sPos[r * PSTR + 64 + c] = (_Float16)v;
    }
    __syncthreads();

    const Seg segNone{nullptr, 0, 0};
    const Seg segPos {sPos,      PSTR, 2};   // K = 64  (pos, padded)
    const Seg segView{sPos + 64, PSTR, 1};   // K = 32  (view, padded)
    const Seg segA   {sA, HSTR, 8};          // K = 256
    const Seg segB   {sB, HSTR, 8};

    // ---- density trunk ----
    layer_std<true>(segPos, segNone, wts + OFF_D1,  64, bd1, sA, mtile, lane);
    layer_std<true>(segA,   segNone, wts + OFF_D2, 256, bd2, sB, mtile, lane);
    layer_std<true>(segB,   segNone, wts + OFF_D3, 256, bd3, sA, mtile, lane);
    layer_std<true>(segA,   segNone, wts + OFF_D4, 256, bd4, sB, mtile, lane);
    layer_std<true>(segPos, segB,    wts + OFF_E1, 320, be1, sA, mtile, lane);  // concat(pos,h)
    layer_std<true>(segA,   segNone, wts + OFF_E2, 256, be2, sB, mtile, lane);
    layer_std<true>(segB,   segNone, wts + OFF_E3, 256, be3, sA, mtile, lane);
    layer_std<true>(segA,   segNone, wts + OFF_E4, 256, be4, sB, mtile, lane);

    // ---- e5: 257 outputs (N padded to 272 = 17 tiles); col 0 = density,
    //      cols 1..256 = feat written to sA cols 0..255. No ReLU. ----
    auto e5_epi = [&](v8f acc, int nt) {
        int n = nt * 16 + m;
        float bv = (n < 257) ? be5[n] : 0.0f;
#pragma unroll
        for (int v = 0; v < 8; ++v) {
            float val = acc[v] + bv;
            int row = mtile * 16 + v + 8 * hi;
            if (n == 0)       sDens[row] = val;
            else if (n < 257) sA[(size_t)row * HSTR + (n - 1)] = (_Float16)val;
        }
    };
    gemm_chunk<8>(segB, segNone, wts + OFF_E5, 256,  0, mtile, lane, e5_epi);
    gemm_chunk<8>(segB, segNone, wts + OFF_E5, 256,  8, mtile, lane, e5_epi);
    gemm_chunk<1>(segB, segNone, wts + OFF_E5, 256, 16, mtile, lane, e5_epi);

    // ---- color head ----
    layer_std<true>(segView, segA, wts + OFF_C1, 288, bc1, sB, mtile, lane);    // concat(view,feat)

    // c2: 3 outputs (1 padded N tile), write rgb straight to global.
    gemm_chunk<1>(segB, segNone, wts + OFF_C2, 256, 0, mtile, lane,
        [&](v8f acc, int nt) {
            (void)nt;
            if (m < 3) {
                float bv = bc2[m];
#pragma unroll
                for (int v = 0; v < 8; ++v) {
                    int row = mtile * 16 + v + 8 * hi;
                    out[(size_t)(row0 + row) * 4 + m] = acc[v] + bv;
                }
            }
        });

    // ---- density channel ----
    __syncthreads();
    for (int r = tid; r < 128; r += 256)
        out[(size_t)(row0 + r) * 4 + 3] = sDens[r];
}

// ------------------------------------------------------------------
extern "C" void kernel_launch(void* const* d_in, const int* in_sizes, int n_in,
                              void* d_out, int out_size, void* d_ws, size_t ws_size,
                              hipStream_t stream)
{
    (void)in_sizes; (void)n_in; (void)out_size; (void)ws_size;
    _Float16* wts = (_Float16*)d_ws;

    auto conv = [&](int idx, int Kin, int Nsrc, size_t off, int Npad, int Kpad,
                    int s0l, int s0p) {
        int total = Npad * Kpad;
        convert_weights<<<(total + 255) / 256, 256, 0, stream>>>(
            (const float*)d_in[idx], Kin, Nsrc, wts + off, Npad, Kpad, s0l, s0p);
    };
    // dict order: x, d1_w,d1_b, d2_w,d2_b, d3_w,d3_b, d4_w,d4_b,
    //             e1_w,e1_b ... e5_w,e5_b, c1_w,c1_b, c2_w,c2_b
    conv( 1,  63, 256, OFF_D1, 256,  64,  63,  64);
    conv( 3, 256, 256, OFF_D2, 256, 256, 256, 256);
    conv( 5, 256, 256, OFF_D3, 256, 256, 256, 256);
    conv( 7, 256, 256, OFF_D4, 256, 256, 256, 256);
    conv( 9, 319, 256, OFF_E1, 256, 320,  63,  64);
    conv(11, 256, 256, OFF_E2, 256, 256, 256, 256);
    conv(13, 256, 256, OFF_E3, 256, 256, 256, 256);
    conv(15, 256, 256, OFF_E4, 256, 256, 256, 256);
    conv(17, 256, 257, OFF_E5, 272, 256, 256, 256);
    conv(19, 283, 256, OFF_C1, 256, 288,  27,  32);
    conv(21, 256,   3, OFF_C2,  16, 256, 256, 256);

    const int nrows  = in_sizes[0] / 90;     // 131072
    const int blocks = nrows / 128;          // 1024
    nerf_fused<<<blocks, 256, SM_BYTES, stream>>>(
        (const float*)d_in[0], wts,
        (const float*)d_in[2],  (const float*)d_in[4],  (const float*)d_in[6],
        (const float*)d_in[8],  (const float*)d_in[10], (const float*)d_in[12],
        (const float*)d_in[14], (const float*)d_in[16], (const float*)d_in[18],
        (const float*)d_in[20], (const float*)d_in[22],
        (float*)d_out);
}
